// ScalarHMM_24781961298348
// MI455X (gfx1250) — matrix operand order repaired and checked by reading, not compiled, not run on hardware
//
#include <hip/hip_runtime.h>
#include <cstddef>

typedef __attribute__((ext_vector_type(16))) _Float16 v16h;
typedef __attribute__((ext_vector_type(8)))  _Float16 v8h;
typedef __attribute__((ext_vector_type(8)))  float    v8f;
typedef __attribute__((ext_vector_type(4)))  float    v4f;

#define HMM_S   128      // states
#define HMM_V   10000    // vocab
#define HMM_T   256      // story length
#define HMM_L   32       // sentence length
#define HMM_MB  16       // stories per workgroup (WMMA M tile)
#define PAD_F   132      // padded f32 LDS row (avoid bank conflicts)
#define PAD_H   136      // padded f16 LDS row (272B -> conflict-free frag loads)

// Raw-hardware transcendentals: inputs here are never subnormal
// (WMMA acc >= ~1e-4), so skip clang's denorm-fixup expansions.
__device__ __forceinline__ float fast_exp(float x) {
  return __builtin_amdgcn_exp2f(x * 1.4426950408889634f);   // v_exp_f32
}
__device__ __forceinline__ float fast_log(float x) {
  return __builtin_amdgcn_logf(x) * 0.6931471805599453f;    // v_log_f32
}

// ---------------------------------------------------------------------------
// Kernel 1: one-time transpose of emissions into EmT[tok][s] (f32, 5.12 MB,
// fully L2-resident on MI455X's 192 MB L2). Makes per-token gathers contiguous.
// ---------------------------------------------------------------------------
__global__ __launch_bounds__(256) void hmm_transpose_emissions(
    const float* __restrict__ em, float* __restrict__ emT) {
  int idx = blockIdx.x * 256 + threadIdx.x;
  if (idx < HMM_V * HMM_S) {
    int tok = idx >> 7;      // idx / 128
    int s   = idx & 127;
    emT[idx] = em[s * HMM_V + tok];   // EmT[tok*128 + s]
  }
}

// ---------------------------------------------------------------------------
// Kernel 2: HMM forward chain. 16 stories / workgroup, 8 waves.
// Wave w owns next-state tile w (16 states). Per step:
//   new[b,i] = em[b,i] + m_b + log( sum_j exp(score[b,j]-m_b) * exp(trans[i,j]) )
// computed as f16 WMMA (A = P tile from LDS, B = exp(trans) frags in VGPRs).
// Emissions for step t+1 are prefetched during step t (double-buffered sEm),
// so the L2 gather latency overlaps a full recursion step. 2 barriers/step.
// ---------------------------------------------------------------------------
__global__ __launch_bounds__(256) void hmm_forward_kernel(
    const float* __restrict__ trans,    // [S,S]  trans[next*S + prev]
    const float* __restrict__ priors,   // [S]
    const int*   __restrict__ stories,  // [B,T,L]
    const float* __restrict__ emT,      // [V,S]
    float*       __restrict__ out)      // [B]
{
  __shared__ float    sScore[HMM_MB][PAD_F];     // running log-scores (f32)
  __shared__ float    sEm[2][HMM_MB][PAD_F];     // emission scores, 2-deep pipe
  __shared__ _Float16 sP[HMM_MB][PAD_H];         // exp(score - rowmax), f16
  __shared__ float    sMb[HMM_MB];               // per-story rescale max

  const int tid = threadIdx.x;
  const int lid = tid & 31;
  const int wv  = tid >> 5;          // wave id == next-state tile (0..7)
  const int m   = tid >> 4;          // story row for gather/reduce (0..15)
  const int sg  = tid & 15;          // 8-state group within row
  const int hi  = lid >> 4;          // lane half
  const int nl  = lid & 15;
  const int b0  = blockIdx.x * HMM_MB;

  // ---- Preload constant B fragments: B[k=prev][n=next] = exp(trans[n][prev])
  // Layout (this code's contiguous k order, A uses the same): lane nl = column N, half h -> K = hi*16+h.
  v16h Bfrag[4];
  {
    const int n = wv * 16 + nl;
    #pragma unroll
    for (int kc = 0; kc < 4; ++kc) {
      const float* tr = &trans[n * HMM_S + kc * 32 + hi * 16];
      v16h tmp;
      #pragma unroll
      for (int h = 0; h < 16; ++h) tmp[h] = (_Float16)fast_exp(tr[h]);
      Bfrag[kc] = tmp;
    }
  }

  // Thread (m, sg): mean emission log-prob of its 8 states over the step's
  // 32 tokens. Tokens read directly (2 cachelines shared by the story's 16
  // threads -> L0 hits); gathers are 2x16B rows from the L2-resident EmT.
  auto gather_em = [&](int t, float ev[8]) {
    const int* tokp = &stories[((size_t)(b0 + m) * HMM_T + t) * HMM_L];
    v4f a0 = {0.f, 0.f, 0.f, 0.f};
    v4f a1 = {0.f, 0.f, 0.f, 0.f};
    float cnt = 0.f;
    #pragma unroll
    for (int c = 0; c < 8; ++c) {
      const int4 tk = ((const int4*)tokp)[c];
      const int tks[4] = {tk.x, tk.y, tk.z, tk.w};
      #pragma unroll
      for (int u = 0; u < 4; ++u) {
        const int tok = tks[u];
        if (tok > -1) {                     // -1 = padding emits 0
          cnt += 1.f;
          const v4f* p = (const v4f*)(emT + (size_t)tok * HMM_S + sg * 8);
          a0 += p[0];
          a1 += p[1];
        }
      }
    }
    const float inv = __builtin_amdgcn_rcpf(cnt);  // exact for cnt = 32
    #pragma unroll
    for (int j = 0; j < 4; ++j) { ev[j] = a0[j] * inv; ev[4 + j] = a1[j] * inv; }
  };

  // ---- prologue: score0 = em(0) + priors; prefetch em(1) into pipe slot 1
  {
    float e0[8];
    gather_em(0, e0);
    const float* pr = &priors[sg * 8];
    #pragma unroll
    for (int j = 0; j < 8; ++j) sScore[m][sg * 8 + j] = e0[j] + pr[j];
    float e1[8];
    gather_em(1, e1);
    #pragma unroll
    for (int j = 0; j < 8; ++j) sEm[1][m][sg * 8 + j] = e1[j];
  }
  __syncthreads();

  // ---- forward chain
  for (int t = 1; t < HMM_T; ++t) {
    // prefetch emissions for t+1 (registers only; latency spans this step)
    float eN[8];
    gather_em((t + 1 < HMM_T) ? t + 1 : HMM_T - 1, eN);

    // per-story max: the 16 reducers of story m are 16 contiguous lanes of
    // one wave -> shuffle reduction, no LDS, no barrier.
    float rm = -3.4e38f;
    #pragma unroll
    for (int j = 0; j < 8; ++j) rm = fmaxf(rm, sScore[m][sg * 8 + j]);
    rm = fmaxf(rm, __shfl_xor(rm, 1, 32));
    rm = fmaxf(rm, __shfl_xor(rm, 2, 32));
    rm = fmaxf(rm, __shfl_xor(rm, 4, 32));
    rm = fmaxf(rm, __shfl_xor(rm, 8, 32));
    if (sg == 0) sMb[m] = rm;
    #pragma unroll
    for (int j = 0; j < 8; ++j)
      sP[m][sg * 8 + j] = (_Float16)fast_exp(sScore[m][sg * 8 + j] - rm);
    __syncthreads();                                    // (B) P ready

    // WMMA: acc[16x16] = P[16x128] * E[128 x 16(tile wv)], K chunks of 32.
    // Two independent accumulator chains interleaved (no D->C stall pairs).
    // A layout (matches B): M = nl; halves 0-15 -> K=hi*16+h:
    // two contiguous 16B LDS loads.
    v16h afrag[4];
    #pragma unroll
    for (int kc = 0; kc < 4; ++kc) {
      const int kb = kc * 32 + hi * 16;
      v8h a0 = *(const v8h*)&sP[nl][kb];
      v8h a1 = *(const v8h*)&sP[nl][kb + 8];
      afrag[kc] = __builtin_shufflevector(a0, a1, 0, 1, 2, 3, 4, 5, 6, 7,
                                                  8, 9, 10, 11, 12, 13, 14, 15);
    }
    v8f accA = {0.f, 0.f, 0.f, 0.f, 0.f, 0.f, 0.f, 0.f};
    v8f accB = {0.f, 0.f, 0.f, 0.f, 0.f, 0.f, 0.f, 0.f};
    accA = __builtin_amdgcn_wmma_f32_16x16x32_f16(false, afrag[0], false,
               Bfrag[0], (short)0, accA, false, false);
    accB = __builtin_amdgcn_wmma_f32_16x16x32_f16(false, afrag[1], false,
               Bfrag[1], (short)0, accB, false, false);
    accA = __builtin_amdgcn_wmma_f32_16x16x32_f16(false, afrag[2], false,
               Bfrag[2], (short)0, accA, false, false);
    accB = __builtin_amdgcn_wmma_f32_16x16x32_f16(false, afrag[3], false,
               Bfrag[3], (short)0, accB, false, false);

    // epilogue: D layout (ISA 7.12.2, 32-bit 16x16): M = r + 8*hi, N = nl.
    // sEm[t&1] was filled one iteration ago; stash eN for the next step.
    const int ng = wv * 16 + nl;
    #pragma unroll
    for (int r = 0; r < 8; ++r) {
      const int M = r + 8 * hi;
      sScore[M][ng] = sEm[t & 1][M][ng] + sMb[M] + fast_log(accA[r] + accB[r]);
    }
    #pragma unroll
    for (int j = 0; j < 8; ++j) sEm[(t + 1) & 1][m][sg * 8 + j] = eN[j];
    __syncthreads();                                    // (D) Score+Em ready
  }

  // ---- finalize: out[b] = logsumexp_s(score[b, s]) via lane shuffles
  float rm = -3.4e38f;
  #pragma unroll
  for (int j = 0; j < 8; ++j) rm = fmaxf(rm, sScore[m][sg * 8 + j]);
  rm = fmaxf(rm, __shfl_xor(rm, 1, 32));
  rm = fmaxf(rm, __shfl_xor(rm, 2, 32));
  rm = fmaxf(rm, __shfl_xor(rm, 4, 32));
  rm = fmaxf(rm, __shfl_xor(rm, 8, 32));
  float ls = 0.f;
  #pragma unroll
  for (int j = 0; j < 8; ++j) ls += fast_exp(sScore[m][sg * 8 + j] - rm);
  ls += __shfl_xor(ls, 1, 32);
  ls += __shfl_xor(ls, 2, 32);
  ls += __shfl_xor(ls, 4, 32);
  ls += __shfl_xor(ls, 8, 32);
  if (sg == 0) out[b0 + m] = rm + fast_log(ls);
}

// ---------------------------------------------------------------------------
extern "C" void kernel_launch(void* const* d_in, const int* in_sizes, int n_in,
                              void* d_out, int out_size, void* d_ws, size_t ws_size,
                              hipStream_t stream) {
  const float* trans   = (const float*)d_in[0];   // [128,128]
  const float* emis    = (const float*)d_in[1];   // [128,10000]
  const float* priors  = (const float*)d_in[2];   // [128]
  const int*   stories = (const int*)d_in[3];     // [B,256,32]
  float* out = (float*)d_out;
  float* emT = (float*)d_ws;                      // needs V*S*4 = 5.12 MB

  const int nB = in_sizes[3] / (HMM_T * HMM_L);   // 1024 stories

  const int nT = HMM_V * HMM_S;
  hmm_transpose_emissions<<<(nT + 255) / 256, 256, 0, stream>>>(emis, emT);
  hmm_forward_kernel<<<nB / HMM_MB, 256, 0, stream>>>(trans, priors, stories,
                                                      emT, out);
}
